// IntervalDistLoss_10015863734567
// MI455X (gfx1250) — matrix operand order repaired and checked by reading, not compiled, not run on hardware
//
#include <hip/hip_runtime.h>
#include <hip/hip_bf16.h>

typedef __attribute__((ext_vector_type(16))) _Float16 v16h;
typedef __attribute__((ext_vector_type(8)))  float    v8f;

#define NS    128   // samples per ray (fixed by reference)
#define WAVES 8     // waves (rays) per block

// Kernel 1: one wave32 per ray. Computes per-block partial of
//   term0_r/2 + term1_r/3   summed over the block's rays.
//
// term0 = 2*(u^T G w - w^T G u), G_ij = [m_i > m_j] in {0,1} (exact in f16),
// u = w.*m. G tiles feed v_wmma_f32_16x16x32_f16; w,u enter as f16 hi+lo
// column pairs so the result keeps fp32-class accuracy.
__global__ __launch_bounds__(256) void distloss_main(
    const float* __restrict__ zv,
    const float* __restrict__ dl,
    const float* __restrict__ wt,
    float* __restrict__ blockOut,
    int R)
{
    __shared__ float    m_s [WAVES][NS];   // mids (f32, exact compares)
    __shared__ float    u_s [WAVES][NS];   // u = w*m      (epilogue, cols 0-1)
    __shared__ float    nw_s[WAVES][NS];   // -w           (epilogue, cols 2-3)
    __shared__ float    zf_s[WAVES][NS];   // zeros        (epilogue, cols >=4)
    __shared__ _Float16 wh_s[WAVES][NS];   // B col 0: f16 hi(w)
    __shared__ _Float16 wl_s[WAVES][NS];   // B col 1: f16 lo(w)
    __shared__ _Float16 uh_s[WAVES][NS];   // B col 2: f16 hi(u)
    __shared__ _Float16 ul_s[WAVES][NS];   // B col 3: f16 lo(u)
    __shared__ _Float16 zh_s[WAVES][NS];   // B cols >=4: zeros
    __shared__ float    wavePart[WAVES];

    const int lane = threadIdx.x & 31;
    const int wid  = threadIdx.x >> 5;
    const int ray  = blockIdx.x * WAVES + wid;
    const bool valid = (ray < R);
    const int  rc  = valid ? ray : (R - 1);

    const float* zr = zv + (size_t)rc * NS;
    const float* dr = dl + (size_t)rc * NS;
    const float* wr = wt + (size_t)rc * NS;

    // ---- Phase 1: load 4 consecutive samples per lane; stage all LDS views ----
    float t1 = 0.0f;
    {
        const int i0 = lane * 4;
        float4 z4 = *(const float4*)(zr + i0);
        float4 d4 = *(const float4*)(dr + i0);
        float4 w4 = *(const float4*)(wr + i0);

        float mm[4] = { z4.x + 0.5f * d4.x, z4.y + 0.5f * d4.y,
                        z4.z + 0.5f * d4.z, z4.w + 0.5f * d4.w };
        float ww[4] = { w4.x, w4.y, w4.z, w4.w };
        float dd[4] = { d4.x, d4.y, d4.z, d4.w };

#pragma unroll
        for (int k = 0; k < 4; ++k) {
            const float uu = ww[k] * mm[k];
            m_s [wid][i0 + k] = mm[k];
            u_s [wid][i0 + k] = uu;
            nw_s[wid][i0 + k] = -ww[k];
            zf_s[wid][i0 + k] = 0.0f;

            const _Float16 whi = (_Float16)ww[k];
            const _Float16 uhi = (_Float16)uu;
            wh_s[wid][i0 + k] = whi;
            wl_s[wid][i0 + k] = (_Float16)(ww[k] - (float)whi);
            uh_s[wid][i0 + k] = uhi;
            ul_s[wid][i0 + k] = (_Float16)(uu - (float)uhi);
            zh_s[wid][i0 + k] = (_Float16)0.0f;

            t1 += ww[k] * ww[k] * dd[k];
        }
    }
    __syncthreads();

    // ---- Phase 2: D = G * [w_hi w_lo u_hi u_lo 0...] via WMMA ----
    v8f acc[8] = {};                       // 8 row tiles of 16 rows, f32 acc

    const int col    = lane & 15;              // B/D column owned by this lane
    const int bKBase = (lane >= 16) ? 16 : 0;  // B: lanes 16-31 hold K=16..31
    const int aKBase = (lane >= 16) ? 16 : 0;  // A: lanes 16-31 hold K=16..31

    // One-time column -> LDS base pointer resolution (no per-element selects)
    const _Float16* bcol =
        (col == 0) ? &wh_s[wid][0] :
        (col == 1) ? &wl_s[wid][0] :
        (col == 2) ? &uh_s[wid][0] :
        (col == 3) ? &ul_s[wid][0] : &zh_s[wid][0];

#pragma unroll
    for (int jc = 0; jc < 4; ++jc) {           // K chunks of 32 (j = 0..127)
        // B build: one contiguous 32B LDS read per lane (layout-contiguous K)
        const v16h b = *(const v16h*)(bcol + jc * 32 + bKBase);

#pragma unroll
        for (int rt = 0; rt < 8; ++rt) {
            const int   row = rt * 16 + (lane & 15);
            const float mi  = m_s[wid][row];
            v16h a;
#pragma unroll
            for (int e = 0; e < 16; ++e) {
                // A layout: K = aKBase+e (contiguous, matches B)
                const int j = jc * 32 + aKBase + e;
                a[e] = (mi > m_s[wid][j]) ? (_Float16)1.0f : (_Float16)0.0f;
            }
            acc[rt] = __builtin_amdgcn_wmma_f32_16x16x32_f16(
                false, a, false, b, (short)0, acc[rt], false, false);
        }
    }

    // ---- Phase 3: contract  u_i*(Gw)_i - w_i*(Gu)_i  (sign folded into -w) ----
    // D vgpr p: lanes 0-15 -> (row p, col=lane); lanes 16-31 -> (row 8+p, col=lane-16)
    float part = t1 * (1.0f / 3.0f);
    const float* cptr = (col < 2) ? &u_s[wid][0]
                      : (col < 4) ? &nw_s[wid][0] : &zf_s[wid][0];
#pragma unroll
    for (int rt = 0; rt < 8; ++rt) {
        const int rbase = rt * 16 + ((lane >= 16) ? 8 : 0);
        float4 c0 = *(const float4*)(cptr + rbase);
        float4 c1 = *(const float4*)(cptr + rbase + 4);
        part += c0.x * acc[rt][0] + c0.y * acc[rt][1]
              + c0.z * acc[rt][2] + c0.w * acc[rt][3]
              + c1.x * acc[rt][4] + c1.y * acc[rt][5]
              + c1.z * acc[rt][6] + c1.w * acc[rt][7];
    }
    if (!valid) part = 0.0f;

    // wave32 reduction
#pragma unroll
    for (int off = 16; off > 0; off >>= 1)
        part += __shfl_xor(part, off, 32);

    if (lane == 0) wavePart[wid] = part;
    __syncthreads();
    if (threadIdx.x == 0) {
        float s = 0.0f;
#pragma unroll
        for (int q = 0; q < WAVES; ++q) s += wavePart[q];
        blockOut[blockIdx.x] = s;
    }
}

// Kernel 2: deterministic final reduction of per-block partials.
__global__ __launch_bounds__(256) void distloss_reduce(
    const float* __restrict__ part, int nb, float invNp, float* __restrict__ out)
{
    __shared__ float s[256];
    float acc = 0.0f;
    for (int i = threadIdx.x; i < nb; i += 256) acc += part[i];
    s[threadIdx.x] = acc;
    __syncthreads();
    for (int st = 128; st > 0; st >>= 1) {
        if ((int)threadIdx.x < st) s[threadIdx.x] += s[threadIdx.x + st];
        __syncthreads();
    }
    if (threadIdx.x == 0) out[0] = 0.01f * s[0] * invNp;
}

extern "C" void kernel_launch(void* const* d_in, const int* in_sizes, int n_in,
                              void* d_out, int out_size, void* d_ws, size_t ws_size,
                              hipStream_t stream) {
    (void)n_in; (void)out_size; (void)ws_size;
    const float* z = (const float*)d_in[0];   // z_vals  [R,128]
    const float* d = (const float*)d_in[1];   // deltas  [R,128]
    const float* w = (const float*)d_in[2];   // weights [R,128]
    const int R  = in_sizes[0] / NS;          // == num_pixels
    const int nb = (R + WAVES - 1) / WAVES;

    float* partials = (float*)d_ws;
    distloss_main<<<nb, 256, 0, stream>>>(z, d, w, partials, R);
    distloss_reduce<<<1, 256, 0, stream>>>(partials, nb, 1.0f / (float)R,
                                           (float*)d_out);
}
